// SAPAModule_26989574488715
// MI455X (gfx1250) — matrix operand order repaired and checked by reading, not compiled, not run on hardware
//
#include <hip/hip_runtime.h>
#include <hip/hip_bf16.h>

typedef __attribute__((ext_vector_type(16))) __bf16 v16bf;
typedef __attribute__((ext_vector_type(8)))  float  v8f;

// ---------------------------------------------------------------------------
// WMMA fragment loaders following CDNA5 ISA 7.12.2 layouts (wave32).
// A-matrix 16x32 bf16: lane m=lane&15 holds row m; VGPR j (pair p):
//   K = (j<4 ? 2j : 16+2(j-4)) + p + 8*(lane>>4)
// B-matrix 32x16 bf16 (KxN): lane n=lane&15 holds col n;
//   K = (j<4 ? 2j : 16+2(j-4)) + p + 8*(lane>>4) -> same as A, so we store B
//   transposed (row = N, contiguous in K) in LDS.
// C/D f32 16x16: lane holds (M = r + 8*(lane>>4), N = lane&15) in v8f[r].
// ---------------------------------------------------------------------------
static __device__ __forceinline__ v16bf load_a_rowptr(const __bf16* rowbase,
                                                      int kbase, int lane) {
  v16bf a;
  const int hf = lane >> 4;
#pragma unroll
  for (int i = 0; i < 16; ++i) {
    const int j = i >> 1, p = i & 1;
    const int kk = (j < 4 ? 2 * j : 16 + 2 * (j - 4)) + p + 8 * hf + kbase;
    a[i] = rowbase[kk];
  }
  return a;
}

static __device__ __forceinline__ v16bf load_b_Trow(const __bf16* nrowbase,
                                                    int kbase, int lane) {
  v16bf b;
  const int hf = lane >> 4;
#pragma unroll
  for (int i = 0; i < 16; ++i) {
    const int j = i >> 1, p = i & 1;
    const int kk = kbase + (j < 4 ? 2 * j : 16 + 2 * (j - 4)) + p + 8 * hf;
    b[i] = nrowbase[kk];
  }
  return b;
}

// ---------------------------------------------------------------------------
// Kernel 1: LayerNorm (over C=256, NCHW input) + projection to E=64 (bf16 out).
// One wave handles 16 spatial positions; 2 waves per block. Single global
// pass: stage raw values as bf16 while accumulating mean/var, normalize in
// LDS in place, then a (16x256)x(256x64) WMMA GEMM with all 4 N-tile
// accumulators held in registers (A-fragment reused across N-tiles).
// Result staged in LDS and written with b128 stores (rows contiguous).
// ---------------------------------------------------------------------------
__global__ __launch_bounds__(64) void ln_proj_kernel(
    const float* __restrict__ in, const float* __restrict__ gamma,
    const float* __restrict__ beta, const float* __restrict__ Wm,
    const float* __restrict__ bias, __bf16* __restrict__ outq, int HW) {
  __shared__ __align__(16) __bf16 Wt[64 * 256];        // Wt[e][c]  (32 KB)
  __shared__ __align__(16) __bf16 stage[2][16 * 256];  // bf16 rows (16 KB)

  const int tid = threadIdx.x;
  const int wave = tid >> 5;
  const int lane = tid & 31;

  // Stage transposed weights so B-fragment K-pairs are contiguous.
  for (int idx = tid; idx < 64 * 256; idx += 64) {
    const int e = idx >> 8, c = idx & 255;
    Wt[idx] = (__bf16)Wm[c * 64 + e];
  }

  const int base = (blockIdx.x * 2 + wave) * 16;  // first of 16 rows
  const int r = lane & 15;                        // row within tile
  const int chalf = lane >> 4;                    // which 128-channel half
  const int gr = base + r;
  const int b = gr / HW;
  const int rem = gr % HW;
  const float* p = in + (size_t)b * 256 * HW + rem;  // channel stride = HW

  // Single global pass: accumulate sum/sumsq, stage raw bf16.
  __bf16* st = stage[wave];
  float s = 0.f, s2 = 0.f;
  const int c0 = chalf * 128;
  for (int c = c0; c < c0 + 128; ++c) {
    const float v = p[(size_t)c * HW];
    s += v;
    s2 += v * v;
    st[r * 256 + c] = (__bf16)v;
  }
  s += __shfl_xor(s, 16, 32);
  s2 += __shfl_xor(s2, 16, 32);
  const float mu = s * (1.f / 256.f);
  const float var = s2 * (1.f / 256.f) - mu * mu;
  const float rstd = rsqrtf(var + 1e-5f);

  // Normalize + affine in place (same-lane cells only).
  for (int c = c0; c < c0 + 128; ++c) {
    const float v = (float)st[r * 256 + c];
    st[r * 256 + c] = (__bf16)((v - mu) * rstd * gamma[c] + beta[c]);
  }
  __syncthreads();  // Wt + all staged rows visible

  // GEMM: (16x256) x (256x64); K in 8 chunks of 32, 4 N-tiles in registers.
  const __bf16* arow = st + (lane & 15) * 256;
  const int hf = lane >> 4;
  v8f accs[4] = {};
#pragma unroll
  for (int kc = 0; kc < 8; ++kc) {
    v16bf a = load_a_rowptr(arow, kc * 32, lane);
#pragma unroll
    for (int nt = 0; nt < 4; ++nt) {
      v16bf bb = load_b_Trow(Wt + (nt * 16 + (lane & 15)) * 256, kc * 32, lane);
      accs[nt] = __builtin_amdgcn_wmma_f32_16x16x32_bf16(
          false, a, false, bb, (short)0, accs[nt], false, false);
    }
  }

  // Epilogue: bias, stage bf16 tile (reuse stage LDS), coalesced b128 writes.
  __bf16* qS = st;  // 16x64 bf16 region (stage no longer needed)
#pragma unroll
  for (int nt = 0; nt < 4; ++nt) {
    const int n = nt * 16 + (lane & 15);
    const float bv = bias[n];
#pragma unroll
    for (int rr = 0; rr < 8; ++rr)
      qS[(rr + 8 * hf) * 64 + n] = (__bf16)(accs[nt][rr] + bv);
  }
  __syncthreads();
  const uint4* srcv = reinterpret_cast<const uint4*>(qS);
  uint4* dstv = reinterpret_cast<uint4*>(outq + (size_t)base * 64);
  for (int t = lane; t < 128; t += 32) dstv[t] = srcv[t];  // 2 KB tile
}

// ---------------------------------------------------------------------------
// Kernel 2: windowed attention upsample.
// One wave per (b, h, 4 consecutive w). The 4 pixels' 5x5 windows union to a
// 5x8 = 40-entry neighborhood, making sim and out dense WMMA GEMMs:
//   sim = Q(16x64) * Ku^T(64x48pad)       (window sel via column shift)
//   out = attn(16x64pad) * V(64pad x 256)
// LDS zero-filled once; OOB neighbors stay zero (= reference's zero-padded
// unfold, since OOB k == 0 post-bias and OOB v == 0). Output tile staged in
// LDS and written back as float2 rows (8 consecutive ow per (c, oh)).
// ---------------------------------------------------------------------------
__global__ __launch_bounds__(32) void sapa_attn_kernel(
    const __bf16* __restrict__ q,  // [B,128,128,64] bf16
    const __bf16* __restrict__ k,  // [B, 64, 64,64] bf16
    const float* __restrict__ x,   // [B,256,64,64] fp32 (NCHW)
    float* __restrict__ out) {     // [B,256,128,128] fp32 (NCHW)
  __shared__ __align__(16) __bf16 ku[48 * 64];    // [n][e]   (6 KB)
  __shared__ __align__(16) __bf16 vT[256 * 64];   // [c][pos] (32 KB)
  __shared__ __align__(16) float simL[16 * 48];   // (3 KB)
  __shared__ __align__(16) __bf16 attn[16 * 64];  // out-GEMM A (2 KB)
  __shared__ __align__(16) float outS[256 * 16];  // [c][m] result (16 KB)

  const int lane = threadIdx.x;
  const int gid = blockIdx.x;  // b * 64 * 16 groups
  const int b = gid / (64 * 16);
  const int hw = gid % (64 * 16);
  const int h = hw / 16;
  const int w0 = (hw % 16) * 4;

  // Zero-fill ku / vT / attn with b128 stores; OOB entries stay zero.
  {
    uint4 z = {0u, 0u, 0u, 0u};
    uint4* zk = reinterpret_cast<uint4*>(ku);
    for (int t = lane; t < (48 * 64 * 2) / 16; t += 32) zk[t] = z;
    uint4* zv = reinterpret_cast<uint4*>(vT);
    for (int t = lane; t < (256 * 64 * 2) / 16; t += 32) zv[t] = z;
    uint4* za = reinterpret_cast<uint4*>(attn);
    for (int t = lane; t < (16 * 64 * 2) / 16; t += 32) za[t] = z;
  }
  __syncthreads();

  // Union neighborhood index: n = dy*8 + dxu, (dy,dxu) in 5x8.
  // K rows are contiguous 128 B -> copy as uint2 (4 bf16).
  for (int t = lane; t < 40 * 16; t += 32) {
    const int n = t >> 4, j = t & 15;
    const int dy = n >> 3, dxu = n & 7;
    const int hh = h - 2 + dy, ww = w0 - 2 + dxu;
    if (hh >= 0 && hh < 64 && ww >= 0 && ww < 64) {
      const uint2* src =
          reinterpret_cast<const uint2*>(k + (((size_t)b * 64 + hh) * 64 + ww) * 64);
      reinterpret_cast<uint2*>(&ku[n * 64])[j] = src[j];
    }
  }
  // V: x rows along w are contiguous; all offsets even -> float2 loads, and
  // pairs are never partially OOB (W=64 even, w0 multiple of 4).
  for (int t = lane; t < 256 * 5 * 4; t += 32) {
    const int c = t / 20;
    const int rest = t % 20;
    const int dy = rest >> 2, j = rest & 3;
    const int hh = h - 2 + dy, ww = w0 - 2 + 2 * j;
    if (hh >= 0 && hh < 64 && ww >= 0 && ww < 64) {
      const float2 v2 = *reinterpret_cast<const float2*>(
          x + (((size_t)b * 256 + c) * 64 + hh) * 64 + ww);
      __bf16* d = &vT[c * 64 + dy * 8 + 2 * j];
      d[0] = (__bf16)v2.x;
      d[1] = (__bf16)v2.y;
    }
  }
  __syncthreads();

  // Query row m = wl*4 + u*2 + v  ->  output pixel (h*2+u, (w0+wl)*2+v).
  const int m = lane & 15, hf = lane >> 4;
  const int wl = m >> 2, u = (m >> 1) & 1, vv = m & 1;
  const int oh = h * 2 + u, ow = (w0 + wl) * 2 + vv;
  const __bf16* qrow = q + (((size_t)b * 128 + oh) * 128 + ow) * 64;

  // sim = Q x Ku^T  (3 N-tiles of 16, K=64 in two chunks).
#pragma unroll
  for (int nt = 0; nt < 3; ++nt) {
    v8f acc = {};
    const __bf16* brow = ku + (nt * 16 + (lane & 15)) * 64;
#pragma unroll
    for (int kc = 0; kc < 2; ++kc) {
      v16bf a = load_a_rowptr(qrow, kc * 32, lane);
      v16bf bb = load_b_Trow(brow, kc * 32, lane);
      acc = __builtin_amdgcn_wmma_f32_16x16x32_bf16(false, a, false, bb,
                                                    (short)0, acc, false, false);
    }
#pragma unroll
    for (int rr = 0; rr < 8; ++rr)
      simL[(rr + 8 * hf) * 48 + nt * 16 + (lane & 15)] = acc[rr];
  }
  __syncthreads();

  // Softmax over each row's 25 window entries; scatter into A with col shift.
  if (lane < 16) {
    float vals[25];
    float mx = -1e30f;
#pragma unroll
    for (int dy = 0; dy < 5; ++dy)
#pragma unroll
      for (int dx = 0; dx < 5; ++dx) {
        const float sv = simL[lane * 48 + dy * 8 + wl + dx];
        vals[dy * 5 + dx] = sv;
        mx = fmaxf(mx, sv);
      }
    float sum = 0.f;
#pragma unroll
    for (int i = 0; i < 25; ++i) {
      vals[i] = __expf(vals[i] - mx);
      sum += vals[i];
    }
    const float inv = 1.f / sum;
#pragma unroll
    for (int dy = 0; dy < 5; ++dy)
#pragma unroll
      for (int dx = 0; dx < 5; ++dx)
        attn[lane * 64 + dy * 8 + wl + dx] = (__bf16)(vals[dy * 5 + dx] * inv);
  }
  __syncthreads();

  // out = attn x V  (16 channel tiles, K=64 in two chunks); stage in LDS.
  const __bf16* arow = attn + (lane & 15) * 64;
#pragma unroll
  for (int ct = 0; ct < 16; ++ct) {
    v8f acc = {};
    const __bf16* brow = vT + (ct * 16 + (lane & 15)) * 64;
#pragma unroll
    for (int kc = 0; kc < 2; ++kc) {
      v16bf a = load_a_rowptr(arow, kc * 32, lane);
      v16bf bb = load_b_Trow(brow, kc * 32, lane);
      acc = __builtin_amdgcn_wmma_f32_16x16x32_bf16(false, a, false, bb,
                                                    (short)0, acc, false, false);
    }
    const int c = ct * 16 + (lane & 15);
#pragma unroll
    for (int rr = 0; rr < 8; ++rr) outS[c * 16 + rr + 8 * hf] = acc[rr];
  }
  __syncthreads();

  // Writeback: per (c, oh) a row of 8 consecutive ow -> float2 stores.
  for (int t = lane; t < 256 * 2 * 4; t += 32) {
    const int c = t >> 3;
    const int u_ = (t >> 2) & 1;
    const int wl_ = t & 3;
    const int m0 = wl_ * 4 + u_ * 2;
    float2 val;
    val.x = outS[c * 16 + m0];
    val.y = outS[c * 16 + m0 + 1];
    *reinterpret_cast<float2*>(out + (((size_t)b * 256 + c) * 128 +
                                      (h * 2 + u_)) * 128 + (w0 + wl_) * 2) = val;
  }
}

// ---------------------------------------------------------------------------
// Launch: y-proj (65536 rows), k-proj (16384 rows), then attention (4096 wgs).
// Workspace: q bf16 (8 MB) + k bf16 (2 MB).
// ---------------------------------------------------------------------------
extern "C" void kernel_launch(void* const* d_in, const int* in_sizes, int n_in,
                              void* d_out, int out_size, void* d_ws,
                              size_t ws_size, hipStream_t stream) {
  const float* y = (const float*)d_in[0];
  const float* x = (const float*)d_in[1];
  const float* gy = (const float*)d_in[2];
  const float* by = (const float*)d_in[3];
  const float* gx = (const float*)d_in[4];
  const float* bx = (const float*)d_in[5];
  const float* Wq = (const float*)d_in[6];
  const float* bq = (const float*)d_in[7];
  const float* Wk = (const float*)d_in[8];
  const float* bk = (const float*)d_in[9];
  float* out = (float*)d_out;

  __bf16* qbuf = (__bf16*)d_ws;              // 65536 * 64 bf16
  __bf16* kbuf = qbuf + (size_t)65536 * 64;  // 16384 * 64 bf16

  // q = LN(y) @ Wq + bq   (B*128*128 = 65536 rows; 32 rows / block)
  ln_proj_kernel<<<2048, 64, 0, stream>>>(y, gy, by, Wq, bq, qbuf, 128 * 128);
  // k = LN(x) @ Wk + bk   (B*64*64 = 16384 rows)
  ln_proj_kernel<<<512, 64, 0, stream>>>(x, gx, bx, Wk, bk, kbuf, 64 * 64);
  // windowed attention upsample: B * H * (W/4) = 4096 one-wave workgroups
  sapa_attn_kernel<<<4096, 32, 0, stream>>>(qbuf, kbuf, x, out);
}
